// MultiHeadAttention_43971875177008
// MI455X (gfx1250) — matrix operand order repaired and checked by reading, not compiled, not run on hardware
//
#include <hip/hip_runtime.h>
#include <hip/hip_bf16.h>

typedef _Float16 v16h __attribute__((ext_vector_type(16)));
typedef _Float16 v8h  __attribute__((ext_vector_type(8)));
typedef float    v8f  __attribute__((ext_vector_type(8)));
typedef float    v4f  __attribute__((ext_vector_type(4)));

#define LOG2E 1.4426950408889634f

#if defined(__has_builtin)
#  if __has_builtin(__builtin_amdgcn_tensor_load_to_lds) && \
      __has_builtin(__builtin_amdgcn_tensor_store_from_lds) && \
      __has_builtin(__builtin_amdgcn_s_wait_tensorcnt)
#    define HAVE_TDM 1
#  endif
#endif
#ifndef HAVE_TDM
#  define HAVE_TDM 0
#endif

static __device__ __forceinline__ v16h make_frag(v8h lo, v8h hi) {
  return __builtin_shufflevector(lo, hi, 0,1,2,3,4,5,6,7,8,9,10,11,12,13,14,15);
}
static __device__ __forceinline__ v8f wmma_f16(v16h a, v16h b, v8f c) {
  // 8 args: (neg_a, A, neg_b, B, c_mod, C, reuse_a, reuse_b)
  return __builtin_amdgcn_wmma_f32_16x16x32_f16(false, a, false, b, (short)0, c, false, false);
}

#if HAVE_TDM
typedef unsigned int u32x4 __attribute__((ext_vector_type(4)));
typedef int          i32x4 __attribute__((ext_vector_type(4)));
typedef int          i32x8 __attribute__((ext_vector_type(8)));

// Build a 2D f16 TDM descriptor (D#):
//   group0: count=1 | lds_addr | global_addr | type=2
//   group1: data_size=1(2B) | tensor_dim0/1 | tile_dim0/1 | dim0_stride
static __device__ __forceinline__ void tdm_build(u32x4& g0, i32x8& g1,
                                                 unsigned lds_off, const void* gp,
                                                 unsigned tile_w, unsigned tile_h,
                                                 unsigned ten_w, unsigned ten_h,
                                                 unsigned stride_elems) {
  unsigned long long ga = (unsigned long long)(uintptr_t)gp;
  g0[0] = 1u;                                   // count=1, user-mode, no gather
  g0[1] = lds_off;                              // LDS byte address
  g0[2] = (unsigned)ga;                         // global_addr[31:0]
  g0[3] = (unsigned)((ga >> 32) & 0x1FFFFFFull) | (2u << 30);  // [56:32] | type=2
  g1[0] = (int)(1u << 16);                      // data_size=1 -> 2 bytes
  g1[1] = (int)((ten_w & 0xFFFFu) << 16);       // tensor_dim0[15:0] @ bits 63:48
  g1[2] = (int)((ten_w >> 16) | ((ten_h & 0xFFFFu) << 16));   // dim0 hi | dim1 lo
  g1[3] = (int)((ten_h >> 16) | (tile_w << 16));              // dim1 hi | tile_dim0
  g1[4] = (int)tile_h;                          // tile_dim1 (tile_dim2 = 0)
  g1[5] = (int)stride_elems;                    // tensor_dim0_stride[31:0]
  g1[6] = 0;                                    // stride hi | dim1_stride lo
  g1[7] = 0;
}
static __device__ __forceinline__ void tdm_load2d(unsigned lds_off, const void* gp,
                                                  unsigned tw, unsigned th,
                                                  unsigned nw, unsigned nh, unsigned st) {
  u32x4 g0; i32x8 g1; tdm_build(g0, g1, lds_off, gp, tw, th, nw, nh, st);
  i32x4 z{};
#if __clang_major__ >= 23
  i32x8 z8{};
  __builtin_amdgcn_tensor_load_to_lds(g0, g1, z, z, z8, 0);
#else
  __builtin_amdgcn_tensor_load_to_lds(g0, g1, z, z, 0);
#endif
}
static __device__ __forceinline__ void tdm_store2d(unsigned lds_off, const void* gp,
                                                   unsigned tw, unsigned th,
                                                   unsigned nw, unsigned nh, unsigned st) {
  u32x4 g0; i32x8 g1; tdm_build(g0, g1, lds_off, gp, tw, th, nw, nh, st);
  i32x4 z{};
#if __clang_major__ >= 23
  i32x8 z8{};
  __builtin_amdgcn_tensor_store_from_lds(g0, g1, z, z, z8, 0);
#else
  __builtin_amdgcn_tensor_store_from_lds(g0, g1, z, z, 0);
#endif
}
#endif  // HAVE_TDM

// ---------------------------------------------------------------------------
// f32 -> f16 conversion pre-pass (bandwidth trivial: ~3.5us of HBM time)
// ---------------------------------------------------------------------------
__global__ __launch_bounds__(256) void cvt_f32_f16(const float* __restrict__ in,
                                                   _Float16* __restrict__ out, int n) {
  const int i = (blockIdx.x * 256 + threadIdx.x) * 8;
  if (i + 8 <= n) {
    v4f f0 = *(const v4f*)(in + i);
    v4f f1 = *(const v4f*)(in + i + 4);
    v8h h;
#pragma unroll
    for (int j = 0; j < 4; ++j) { h[j] = (_Float16)f0[j]; h[j + 4] = (_Float16)f1[j]; }
    *(v8h*)(out + i) = h;
  }
}

// ---------------------------------------------------------------------------
// C = A(8192x1024) * B^T, A/B f16, B row-major (N=1024, K) (Linear layout).
// WG 256 thr (8 waves), 128x128 tile, K-step 32, TDM double-buffered staging.
// MODE 0: f16 row-major out via LDS + tensor_store_from_lds   (Q, K GEMMs)
// MODE 1: f16 transposed out (b*1024+n, s), packed b128       (V^T)
// MODE 2: f32 row-major out                                   (final projection)
// ---------------------------------------------------------------------------
template<int MODE>
__global__ __launch_bounds__(256) void gemm128(const _Float16* __restrict__ A,
                                               const _Float16* __restrict__ B,
                                               void* __restrict__ Out) {
  union SMem {
    _Float16 ab[2][2][128][32];  // [A/B][double-buffer][row][k]
    _Float16 c[128][128];        // epilogue C tile (overlays ab)
  };
  __shared__ alignas(16) SMem sm;

  const int tid  = threadIdx.x;
  const int lane = tid & 31;
  const int wv   = tid >> 5;
  const int wm   = (wv & 3) * 32;
  const int wn   = (wv >> 2) * 64;
  const int mblock = blockIdx.y * 128;
  const int nblock = blockIdx.x * 128;
  const int hh = lane >> 4;
  const int lm = lane & 15;
#if !HAVE_TDM
  const int lrow = tid >> 1;
  const int lcol = (tid & 1) * 16;
#endif

  v8f acc[2][4];
#pragma unroll
  for (int i = 0; i < 2; ++i)
#pragma unroll
    for (int t = 0; t < 4; ++t)
#pragma unroll
      for (int r = 0; r < 8; ++r) acc[i][t][r] = 0.0f;

#if HAVE_TDM
  if (wv == 0) {  // prologue: stage k-tile 0 into buffer 0
    tdm_load2d((unsigned)(uintptr_t)&sm.ab[0][0][0][0],
               A + (size_t)mblock * 1024, 32, 128, 1024, 8192, 1024);
    tdm_load2d((unsigned)(uintptr_t)&sm.ab[1][0][0][0],
               B + (size_t)nblock * 1024, 32, 128, 1024, 1024, 1024);
  }
#endif

  for (int it = 0; it < 32; ++it) {
    const int k0  = it * 32;
    const int buf = it & 1;
#if HAVE_TDM
    if (wv == 0) {
      if (it + 1 < 32) {  // stage next tile into the other buffer, keep 2 in flight
        tdm_load2d((unsigned)(uintptr_t)&sm.ab[0][buf ^ 1][0][0],
                   A + (size_t)mblock * 1024 + k0 + 32, 32, 128, 1024, 8192, 1024);
        tdm_load2d((unsigned)(uintptr_t)&sm.ab[1][buf ^ 1][0][0],
                   B + (size_t)nblock * 1024 + k0 + 32, 32, 128, 1024, 1024, 1024);
        __builtin_amdgcn_s_wait_tensorcnt(2);
      } else {
        __builtin_amdgcn_s_wait_tensorcnt(0);
      }
    }
    __syncthreads();
#else
    {  // fallback: manual f16 staging
      const _Float16* a = A + (size_t)(mblock + lrow) * 1024 + k0 + lcol;
      const _Float16* b = B + (size_t)(nblock + lrow) * 1024 + k0 + lcol;
      *(v8h*)&sm.ab[0][buf][lrow][lcol]     = *(const v8h*)a;
      *(v8h*)&sm.ab[0][buf][lrow][lcol + 8] = *(const v8h*)(a + 8);
      *(v8h*)&sm.ab[1][buf][lrow][lcol]     = *(const v8h*)b;
      *(v8h*)&sm.ab[1][buf][lrow][lcol + 8] = *(const v8h*)(b + 8);
    }
    __syncthreads();
#endif
    // A fragments: lane m=lm, halves {h*8.., h*8+16..}
    v16h a0 = make_frag(*(const v8h*)&sm.ab[0][buf][wm + lm][hh * 8],
                        *(const v8h*)&sm.ab[0][buf][wm + lm][hh * 8 + 16]);
    v16h a1 = make_frag(*(const v8h*)&sm.ab[0][buf][wm + 16 + lm][hh * 8],
                        *(const v8h*)&sm.ab[0][buf][wm + 16 + lm][hh * 8 + 16]);
    // B fragments: lane n=lm, halves {h*8.., h*8+16..}
    v16h bf[4];
#pragma unroll
    for (int t = 0; t < 4; ++t) {
      const _Float16* bp = &sm.ab[1][buf][wn + t * 16 + lm][hh * 8];
      bf[t] = make_frag(*(const v8h*)bp, *(const v8h*)(bp + 16));
    }
#pragma unroll
    for (int t = 0; t < 4; ++t) {
      acc[0][t] = wmma_f16(a0, bf[t], acc[0][t]);
      acc[1][t] = wmma_f16(a1, bf[t], acc[1][t]);
    }
    __syncthreads();
  }

  // ---- epilogue ----
  if (MODE == 0) {
#if HAVE_TDM
    // write C tile to LDS, then one TDM store of the whole 128x128 f16 tile
#pragma unroll
    for (int i = 0; i < 2; ++i)
#pragma unroll
      for (int t = 0; t < 4; ++t)
#pragma unroll
        for (int r = 0; r < 8; ++r)
          sm.c[wm + i * 16 + hh * 8 + r][wn + t * 16 + lm] = (_Float16)acc[i][t][r];
    __syncthreads();
    if (wv == 0)
      tdm_store2d((unsigned)(uintptr_t)&sm.c[0][0],
                  (_Float16*)Out + (size_t)mblock * 1024 + nblock,
                  128, 128, 1024, 8192, 1024);
    // S_ENDPGM's implicit wait-idle covers TENSORcnt
#else
#pragma unroll
    for (int i = 0; i < 2; ++i)
#pragma unroll
      for (int t = 0; t < 4; ++t) {
        const int n  = nblock + wn + t * 16 + lm;
        const int mb = mblock + wm + i * 16 + hh * 8;
        _Float16* o = (_Float16*)Out;
#pragma unroll
        for (int r = 0; r < 8; ++r)
          o[(size_t)(mb + r) * 1024 + n] = (_Float16)acc[i][t][r];
      }
#endif
  } else {
#pragma unroll
    for (int i = 0; i < 2; ++i)
#pragma unroll
      for (int t = 0; t < 4; ++t) {
        const int n  = nblock + wn + t * 16 + lm;
        const int mb = mblock + wm + i * 16 + hh * 8;  // 8 consecutive rows
        if (MODE == 2) {
          float* o = (float*)Out;
#pragma unroll
          for (int r = 0; r < 8; ++r)
            o[(size_t)(mb + r) * 1024 + n] = acc[i][t][r];
        } else {  // MODE 1: V^T (b*1024+n, s); rows contiguous in s -> b128
          const int b = mb / 2048;
          const int s = mb % 2048;
          _Float16* o = (_Float16*)Out + ((size_t)(b * 1024 + n)) * 2048 + s;
          v8h pk;
#pragma unroll
          for (int r = 0; r < 8; ++r) pk[r] = (_Float16)acc[i][t][r];
          *(v8h*)o = pk;
        }
      }
  }
}

// ---------------------------------------------------------------------------
// Flash attention: grid (S/64, H, B), block 128 (4 waves), key blocks of 32.
// Row-sum of P computed as P x ones via WMMA (replaces shuffle-sum butterfly).
// ---------------------------------------------------------------------------
__global__ __launch_bounds__(128) void attn_flash(const _Float16* __restrict__ Q,
                                                  const _Float16* __restrict__ K,
                                                  const _Float16* __restrict__ Vt,
                                                  _Float16* __restrict__ Ctx) {
  __shared__ alignas(16) _Float16 Pst[4][16][32];  // wave-private P re-layout patch

  const int lane = threadIdx.x & 31;
  const int wv   = threadIdx.x >> 5;
  const int hh   = lane >> 4;
  const int lm   = lane & 15;
  const int b    = blockIdx.z;
  const int h    = blockIdx.y;
  const int q0   = blockIdx.x * 64 + wv * 16;

  const _Float16* qbase = Q + ((size_t)(b * 2048 + q0 + lm)) * 1024 + h * 64;
  v16h qa[2];
#pragma unroll
  for (int c = 0; c < 2; ++c) {
    const _Float16* p = qbase + c * 32 + hh * 8;
    qa[c] = make_frag(*(const v8h*)p, *(const v8h*)(p + 16));
  }

  v16h onesf;
#pragma unroll
  for (int i = 0; i < 16; ++i) onesf[i] = (_Float16)1.0f;

  v8f acc[4], lacc;
#pragma unroll
  for (int t = 0; t < 4; ++t)
#pragma unroll
    for (int r = 0; r < 8; ++r) acc[t][r] = 0.0f;
#pragma unroll
  for (int r = 0; r < 8; ++r) lacc[r] = 0.0f;
  float mrow[8];
#pragma unroll
  for (int r = 0; r < 8; ++r) mrow[r] = -1e30f;

  const int nkb = (q0 + 15) / 32 + 1;  // causal trip count
  for (int kb = 0; kb < nkb; ++kb) {
    const int key0 = kb * 32;

    v8f sc[2];
#pragma unroll
    for (int nt = 0; nt < 2; ++nt) {
#pragma unroll
      for (int r = 0; r < 8; ++r) sc[nt][r] = 0.0f;
      const _Float16* kp = K + ((size_t)(b * 2048 + key0 + nt * 16 + lm)) * 1024 + h * 64;
#pragma unroll
      for (int c = 0; c < 2; ++c) {
        const _Float16* p = kp + c * 32 + hh * 8;
        v16h kf = make_frag(*(const v8h*)p, *(const v8h*)(p + 16));
        sc[nt] = wmma_f16(qa[c], kf, sc[nt]);
      }
    }

    const bool needMask = (key0 + 31 > q0);
#pragma unroll
    for (int r = 0; r < 8; ++r) {
      const int qrow = q0 + r + hh * 8;
      float s0 = sc[0][r] * 0.125f;  // 1/sqrt(64)
      float s1 = sc[1][r] * 0.125f;
      if (needMask) {
        if (key0 + lm      > qrow) s0 = -1e30f;
        if (key0 + 16 + lm > qrow) s1 = -1e30f;
      }
      float rm = fmaxf(s0, s1);
#pragma unroll
      for (int d = 1; d < 16; d <<= 1) rm = fmaxf(rm, __shfl_xor(rm, d, 32));
      const float mn    = fmaxf(mrow[r], rm);
      const float alpha = exp2f((mrow[r] - mn) * LOG2E);
      const float p0 = exp2f((s0 - mn) * LOG2E);
      const float p1 = exp2f((s1 - mn) * LOG2E);
      mrow[r] = mn;
      lacc[r] *= alpha;
#pragma unroll
      for (int t = 0; t < 4; ++t) acc[t][r] *= alpha;
      const int prow = r + hh * 8;
      Pst[wv][prow][lm]      = (_Float16)p0;
      Pst[wv][prow][16 + lm] = (_Float16)p1;
    }

    const _Float16* pp = &Pst[wv][lm][hh * 8];
    v16h pa = make_frag(*(const v8h*)pp, *(const v8h*)(pp + 16));
    lacc = wmma_f16(pa, onesf, lacc);  // row-sum of P via matrix core
#pragma unroll
    for (int t = 0; t < 4; ++t) {
      const _Float16* vp =
          Vt + ((size_t)(b * 1024 + h * 64 + t * 16 + lm)) * 2048 + key0 + hh * 8;
      v16h vf = make_frag(*(const v8h*)vp, *(const v8h*)(vp + 16));
      acc[t] = wmma_f16(pa, vf, acc[t]);
    }
  }

#pragma unroll
  for (int t = 0; t < 4; ++t) {
#pragma unroll
    for (int r = 0; r < 8; ++r) {
      const float o = acc[t][r] / lacc[r];
      Ctx[((size_t)(b * 2048 + q0 + r + hh * 8)) * 1024 + h * 64 + t * 16 + lm] =
          (_Float16)o;
    }
  }
}

extern "C" void kernel_launch(void* const* d_in, const int* in_sizes, int n_in,
                              void* d_out, int out_size, void* d_ws, size_t ws_size,
                              hipStream_t stream) {
  const float* x  = (const float*)d_in[0];
  const float* Wq = (const float*)d_in[1];
  const float* Wk = (const float*)d_in[2];
  const float* Wv = (const float*)d_in[3];
  const float* Wo = (const float*)d_in[4];

  const int NX = 8192 * 1024;   // x / Q / K / V / ctx element count
  const int NW = 1024 * 1024;   // weight element count

  _Float16* p    = (_Float16*)d_ws;
  _Float16* x16  = p; p += NX;
  _Float16* Wq16 = p; p += NW;
  _Float16* Wk16 = p; p += NW;
  _Float16* Wv16 = p; p += NW;
  _Float16* Wo16 = p; p += NW;
  _Float16* Q16  = p; p += NX;
  _Float16* K16  = p; p += NX;
  _Float16* Vt16 = p; p += NX;
  _Float16* C16  = p;           // total ~92 MB workspace

  cvt_f32_f16<<<NX / 2048, 256, 0, stream>>>(x,  x16,  NX);
  cvt_f32_f16<<<NW / 2048, 256, 0, stream>>>(Wq, Wq16, NW);
  cvt_f32_f16<<<NW / 2048, 256, 0, stream>>>(Wk, Wk16, NW);
  cvt_f32_f16<<<NW / 2048, 256, 0, stream>>>(Wv, Wv16, NW);
  cvt_f32_f16<<<NW / 2048, 256, 0, stream>>>(Wo, Wo16, NW);

  const dim3 gg(8, 64);
  const dim3 gb(256);
  gemm128<0><<<gg, gb, 0, stream>>>(x16, Wq16, (void*)Q16);
  gemm128<0><<<gg, gb, 0, stream>>>(x16, Wk16, (void*)K16);
  gemm128<1><<<gg, gb, 0, stream>>>(x16, Wv16, (void*)Vt16);

  attn_flash<<<dim3(32, 16, 4), dim3(128), 0, stream>>>(Q16, K16, Vt16, C16);

  gemm128<2><<<gg, gb, 0, stream>>>(C16, Wo16, d_out);
}
